// MVRL_GCN_21732534518446
// MI455X (gfx1250) — matrix operand order repaired and checked by reading, not compiled, not run on hardware
//
#include <hip/hip_runtime.h>
#include <hip/hip_bf16.h>

// B=64, N=1024, IN_DIM=128, HID=32
// d_ws layout: h   : [64][1024][32] _Float16  (4 MB)
//              hWT : [64][32][1024] _Float16  (4 MB)   -> needs ws_size >= 8 MB

typedef __attribute__((ext_vector_type(16))) _Float16 v16h;
typedef __attribute__((ext_vector_type(8)))  _Float16 v8h;
typedef __attribute__((ext_vector_type(8)))  float    v8f;
typedef __attribute__((ext_vector_type(4)))  float    v4f;

// Load a 16x32 f32 tile row into a WMMA 16-bit A operand in this file's contiguous k order (B matches),
// converting f32->f16. rowPtr points at row (n0 + lane&15), column m0.
// kh = lane>>4. Elements 0..7  <- K = 16*kh + 0..7
//               Elements 8..15 <- K = 16*kh + 8..15
__device__ __forceinline__ v16h load_a_f32(const float* __restrict__ rowPtr, int kh) {
    const float* p1 = rowPtr + 16 * kh;
    const float* p2 = p1 + 8;
    v4f q0 = *(const v4f*)(p1);
    v4f q1 = *(const v4f*)(p1 + 4);
    v4f q2 = *(const v4f*)(p2);
    v4f q3 = *(const v4f*)(p2 + 4);
    v16h a;
#pragma unroll
    for (int i = 0; i < 4; ++i) {
        a[i]      = (_Float16)q0[i];
        a[4 + i]  = (_Float16)q1[i];
        a[8 + i]  = (_Float16)q2[i];
        a[12 + i] = (_Float16)q3[i];
    }
    return a;
}

// ---------------- Layer-1 linear: hWT = (x @ W1)^T, F=128 ----------------
__global__ __launch_bounds__(256) void k_linear1(const float* __restrict__ x,
                                                 const float* __restrict__ W,
                                                 _Float16* __restrict__ hWT) {
    __shared__ _Float16 Wt[32][128];   // Wt[o][f] = W[f][o], f16
    const int b = blockIdx.y;
    for (int i = threadIdx.x; i < 128 * 32; i += 256) {
        int f = i >> 5, o = i & 31;
        Wt[o][f] = (_Float16)W[i];
    }
    __syncthreads();

    const int l  = threadIdx.x & 31;
    const int w  = threadIdx.x >> 5;
    const int lo = l & 15;
    const int kh = l >> 4;
    const int n0 = (blockIdx.x * 8 + w) * 16;

    const float* xrow = x + ((size_t)b * 1024 + n0 + lo) * 128;
    v8f c0 = {}, c1 = {};
#pragma unroll
    for (int f0 = 0; f0 < 128; f0 += 32) {
        v16h a  = load_a_f32(xrow + f0, kh);
        v16h b0 = *(const v16h*)&Wt[lo][f0 + 16 * kh];
        v16h b1 = *(const v16h*)&Wt[16 + lo][f0 + 16 * kh];
        c0 = __builtin_amdgcn_wmma_f32_16x16x32_f16(false, a, false, b0, (short)0, c0, false, false);
        c1 = __builtin_amdgcn_wmma_f32_16x16x32_f16(false, a, false, b1, (short)0, c1, false, false);
    }
    // D element r on lane l = D[M = r + 8*kh][N = lo]; store transposed: hWT[b][o][m]
    const int mbase = n0 + 8 * kh;
    v8h p0, p1;
#pragma unroll
    for (int r = 0; r < 8; ++r) { p0[r] = (_Float16)c0[r]; p1[r] = (_Float16)c1[r]; }
    *(v8h*)(hWT + ((size_t)b * 32 + lo) * 1024 + mbase)      = p0;
    *(v8h*)(hWT + ((size_t)b * 32 + 16 + lo) * 1024 + mbase) = p1;
}

// ---------------- Hidden linear: hWT = (h @ W)^T, F=32 ----------------
__global__ __launch_bounds__(256) void k_linearH(const _Float16* __restrict__ h,
                                                 const float* __restrict__ W,
                                                 _Float16* __restrict__ hWT) {
    __shared__ _Float16 Wt[32][32];
    const int b = blockIdx.y;
    for (int i = threadIdx.x; i < 32 * 32; i += 256) {
        int f = i >> 5, o = i & 31;
        Wt[o][f] = (_Float16)W[i];
    }
    __syncthreads();

    const int l  = threadIdx.x & 31;
    const int w  = threadIdx.x >> 5;
    const int lo = l & 15;
    const int kh = l >> 4;
    const int n0 = (blockIdx.x * 8 + w) * 16;

    const _Float16* hrow = h + ((size_t)b * 1024 + n0 + lo) * 32;
    v8h r1 = *(const v8h*)(hrow + 16 * kh);
    v8h r2 = *(const v8h*)(hrow + 16 * kh + 8);
    v16h a;
#pragma unroll
    for (int i = 0; i < 8; ++i) { a[i] = r1[i]; a[8 + i] = r2[i]; }

    v16h b0 = *(const v16h*)&Wt[lo][16 * kh];
    v16h b1 = *(const v16h*)&Wt[16 + lo][16 * kh];
    v8f c0 = {}, c1 = {};
    c0 = __builtin_amdgcn_wmma_f32_16x16x32_f16(false, a, false, b0, (short)0, c0, false, false);
    c1 = __builtin_amdgcn_wmma_f32_16x16x32_f16(false, a, false, b1, (short)0, c1, false, false);

    const int mbase = n0 + 8 * kh;
    v8h p0, p1;
#pragma unroll
    for (int r = 0; r < 8; ++r) { p0[r] = (_Float16)c0[r]; p1[r] = (_Float16)c1[r]; }
    *(v8h*)(hWT + ((size_t)b * 32 + lo) * 1024 + mbase)      = p0;
    *(v8h*)(hWT + ((size_t)b * 32 + 16 + lo) * 1024 + mbase) = p1;
}

// ------- Aggregation: h = PReLU(adj @ hW + bias); col-sums -> out -------
// hWT for this batch (64 KB) is staged into LDS once per block; B-operand
// fetches become ds_load_b128, cutting per-block L2 traffic ~8x. adj streams
// straight from global (each element read exactly once per layer).
__global__ __launch_bounds__(256) void k_aggregate(const float* __restrict__ adj,
                                                   const _Float16* __restrict__ hWT,
                                                   const float* __restrict__ bias,
                                                   const float* __restrict__ alpha,
                                                   _Float16* __restrict__ h,
                                                   float* __restrict__ outp) {
    __shared__ _Float16 sW[32][1024];   // hWT[b][o][m], 64 KB
    __shared__ float lsum[32];
    const int b = blockIdx.y;
    if (threadIdx.x < 32) lsum[threadIdx.x] = 0.f;
    {   // cooperative 64 KB copy, 16-byte granules
        const v8h* src = (const v8h*)(hWT + (size_t)b * 32 * 1024);
        v8h* dst = (v8h*)&sW[0][0];
#pragma unroll
        for (int i = 0; i < 16; ++i) dst[threadIdx.x + 256 * i] = src[threadIdx.x + 256 * i];
    }
    __syncthreads();

    const int l  = threadIdx.x & 31;
    const int w  = threadIdx.x >> 5;
    const int lo = l & 15;
    const int kh = l >> 4;
    const int n0 = (blockIdx.x * 8 + w) * 16;

    const float av = alpha[0];
    const float bias0 = bias[lo];
    const float bias1 = bias[16 + lo];

    const float* adjrow = adj + ((size_t)b * 1024 + n0 + lo) * 1024;

    v8f c0 = {}, c1 = {};
#pragma unroll 4
    for (int m0 = 0; m0 < 1024; m0 += 32) {
        __builtin_prefetch(adjrow + m0 + 128, 0, 0);
        v16h A  = load_a_f32(adjrow + m0, kh);
        v16h B0 = *(const v16h*)&sW[lo][m0 + 16 * kh];
        v16h B1 = *(const v16h*)&sW[16 + lo][m0 + 16 * kh];
        c0 = __builtin_amdgcn_wmma_f32_16x16x32_f16(false, A, false, B0, (short)0, c0, false, false);
        c1 = __builtin_amdgcn_wmma_f32_16x16x32_f16(false, A, false, B1, (short)0, c1, false, false);
    }

    // Epilogue: bias + PReLU, store h (row-major f16), accumulate column sums
    float s0 = 0.f, s1 = 0.f;
    _Float16* hbase = h + (size_t)b * 1024 * 32;
#pragma unroll
    for (int r = 0; r < 8; ++r) {
        const int m = n0 + 8 * kh + r;
        float v0 = c0[r] + bias0; v0 = (v0 >= 0.f) ? v0 : av * v0;
        float v1 = c1[r] + bias1; v1 = (v1 >= 0.f) ? v1 : av * v1;
        s0 += v0; s1 += v1;
        hbase[(size_t)m * 32 + lo]      = (_Float16)v0;
        hbase[(size_t)m * 32 + 16 + lo] = (_Float16)v1;
    }
    atomicAdd(&lsum[lo], s0);
    atomicAdd(&lsum[16 + lo], s1);
    __syncthreads();
    if (threadIdx.x < 32) atomicAdd(&outp[(size_t)b * 128 + threadIdx.x], lsum[threadIdx.x]);
}

__global__ void k_zero(float* __restrict__ p, int n) {
    int i = blockIdx.x * 256 + threadIdx.x;
    if (i < n) p[i] = 0.f;
}

extern "C" void kernel_launch(void* const* d_in, const int* in_sizes, int n_in,
                              void* d_out, int out_size, void* d_ws, size_t ws_size,
                              hipStream_t stream) {
    const float* x   = (const float*)d_in[0];
    const float* adj = (const float*)d_in[1];
    const float* Wl[4] = {(const float*)d_in[2], (const float*)d_in[3],
                          (const float*)d_in[4], (const float*)d_in[5]};
    const float* bl[4] = {(const float*)d_in[6], (const float*)d_in[7],
                          (const float*)d_in[8], (const float*)d_in[9]};
    const float* al[4] = {(const float*)d_in[10], (const float*)d_in[11],
                          (const float*)d_in[12], (const float*)d_in[13]};
    float* out = (float*)d_out;

    _Float16* h   = (_Float16*)d_ws;                 // [64][1024][32]
    _Float16* hWT = h + (size_t)64 * 1024 * 32;      // [64][32][1024]

    dim3 grid(8, 64), blk(256);
    k_zero<<<(64 * 128 + 255) / 256, 256, 0, stream>>>(out, 64 * 128);

    k_linear1<<<grid, blk, 0, stream>>>(x, Wl[0], hWT);
    k_aggregate<<<grid, blk, 0, stream>>>(adj, hWT, bl[0], al[0], h, out + 0);
    for (int layer = 1; layer < 4; ++layer) {
        k_linearH<<<grid, blk, 0, stream>>>(h, Wl[layer], hWT);
        k_aggregate<<<grid, blk, 0, stream>>>(adj, hWT, bl[layer], al[layer], h, out + layer * 32);
    }
}
